// ImprovedGNN_44272522887704
// MI455X (gfx1250) — matrix operand order repaired and checked by reading, not compiled, not run on hardware
//
#include <hip/hip_runtime.h>
#include <math.h>

// ---------------------------------------------------------------------------
// GNN pipeline for MI455X (gfx1250, wave32, WMMA).
// Dense linear layers -> v_wmma_f32_16x16x32_bf16 (fp32 accumulate).
// Sparse aggregation  -> global f32 atomics (L2-resident working set).
// Pre-BN biases dropped (mathematically cancel inside BatchNorm).
// GEMM inner loop: issue all fragment loads per K-step before the 4 WMMAs so
// the scheduler can use partial s_wait_loadcnt and overlap loads with matrix
// math instead of s_wait_loadcnt 0 before every WMMA.
// ---------------------------------------------------------------------------

#define NNODES 50000
#define NEDGES 400000

typedef __attribute__((ext_vector_type(16))) __bf16 v16bf;
typedef __attribute__((ext_vector_type(8)))  float  v8f;

struct alignas(16) U4 { unsigned x, y, z, w; };
struct U4x2 { U4 a, b; };

__device__ __forceinline__ v16bf load_frag(const unsigned short* p0,
                                           const unsigned short* p1) {
  U4x2 f;
  f.a = *(const U4*)p0;
  f.b = *(const U4*)p1;
  return __builtin_bit_cast(v16bf, f);
}

__device__ __forceinline__ void atomAddF(float* p, float v) {
  unsafeAtomicAdd(p, v);   // global_atomic_add_f32
}

__device__ __forceinline__ void atomMaxF(float* addr, float val) {
  // float-ordering == int-ordering for >=0, reversed uint-ordering for <0
  if (val >= 0.0f) atomicMax((int*)addr, __float_as_int(val));
  else             atomicMin((unsigned int*)addr, __float_as_uint(val));
}

// --------------------------- dense GEMM (WMMA) -----------------------------
// C[M x Ncols] = A[M x K](bf16) @ W[Ncols x K](bf16)^T ; fp32 accumulate.
// One wave computes a 16x64 output block with 4 accumulators.
__global__ void __launch_bounds__(256)
k_gemm_bf16(const unsigned short* __restrict__ A,
            const unsigned short* __restrict__ W,
            float* __restrict__ C,
            int K, int Ncols, int total_tiles, int nt64)
{
  const int wave = threadIdx.x >> 5;
  const int tid  = blockIdx.x * 8 + wave;
  if (tid >= total_tiles) return;           // uniform per wave: EXEC stays all-1
  const int mtile = tid / nt64;
  const int nblk  = tid - mtile * nt64;
  const int lane  = threadIdx.x & 31;
  const int half  = lane >> 4;
  const int r     = lane & 15;

  const unsigned short* Abase = A + (size_t)(mtile * 16 + r) * K;
  const int cb = nblk * 64;
  // per-tile weight row pointers (column cb + t*16 + r)
  const unsigned short* Wp0 = W + (size_t)(cb +  0 + r) * K + half * 16;
  const unsigned short* Wp1 = W + (size_t)(cb + 16 + r) * K + half * 16;
  const unsigned short* Wp2 = W + (size_t)(cb + 32 + r) * K + half * 16;
  const unsigned short* Wp3 = W + (size_t)(cb + 48 + r) * K + half * 16;

  v8f acc[4] = {};
  for (int k0 = 0; k0 < K; k0 += 32) {
    // Issue ALL independent fragment loads first (10x global_load_b128),
    // then run the 4 WMMAs -> scheduler can overlap loads with matrix ops.
    v16bf af = load_frag(Abase + k0 + half * 16, Abase + k0 + half * 16 + 8);
    v16bf b0 = load_frag(Wp0 + k0, Wp0 + k0 + 8);
    v16bf b1 = load_frag(Wp1 + k0, Wp1 + k0 + 8);
    v16bf b2 = load_frag(Wp2 + k0, Wp2 + k0 + 8);
    v16bf b3 = load_frag(Wp3 + k0, Wp3 + k0 + 8);
    acc[0] = __builtin_amdgcn_wmma_f32_16x16x32_bf16(false, af, false, b0, (short)0, acc[0], false, false);
    acc[1] = __builtin_amdgcn_wmma_f32_16x16x32_bf16(false, af, false, b1, (short)0, acc[1], false, false);
    acc[2] = __builtin_amdgcn_wmma_f32_16x16x32_bf16(false, af, false, b2, (short)0, acc[2], false, false);
    acc[3] = __builtin_amdgcn_wmma_f32_16x16x32_bf16(false, af, false, b3, (short)0, acc[3], false, false);
  }
#pragma unroll
  for (int t = 0; t < 4; ++t) {
    const int col = cb + t * 16 + r;
#pragma unroll
    for (int v = 0; v < 8; ++v) {
      const int row = mtile * 16 + half * 8 + v;
      C[(size_t)row * Ncols + col] = acc[t][v];
    }
  }
}

// ------------------------------ elementwise --------------------------------
__global__ void k_cvt_bf16(const float* __restrict__ in,
                           unsigned short* __restrict__ out, int n) {
  int i = blockIdx.x * blockDim.x + threadIdx.x;
  if (i < n) {
    __bf16 v = (__bf16)in[i];
    out[i] = __builtin_bit_cast(unsigned short, v);
  }
}

__global__ void k_fill_f32(float* p, float v, int n) {
  int i = blockIdx.x * blockDim.x + threadIdx.x;
  if (i < n) p[i] = v;
}

// ------------------------------ graph ops ----------------------------------
__global__ void k_deg(const int* __restrict__ dst, float* __restrict__ deg) {
  int e = blockIdx.x * blockDim.x + threadIdx.x;
  if (e < NEDGES) atomAddF(&deg[dst[e]], 1.0f);
}

// msg[dst] += xl[src], 128 features, one wave per edge
__global__ void k_sage_agg(const int* __restrict__ src, const int* __restrict__ dst,
                           const float* __restrict__ xl, float* __restrict__ msg) {
  int e = blockIdx.x * 8 + (threadIdx.x >> 5);
  if (e >= NEDGES) return;
  int lane = threadIdx.x & 31;
  const float* xr = xl + (size_t)src[e] * 128;
  float* m = msg + (size_t)dst[e] * 128;
#pragma unroll
  for (int j = 0; j < 4; ++j) {
    int f = lane + j * 32;
    atomAddF(&m[f], xr[f]);
  }
}

// in-place: msg = msg/max(deg,1) + xr     (bias dropped: cancels in BN)
__global__ void k_sage_combine(float* __restrict__ msg, const float* __restrict__ xr,
                               const float* __restrict__ deg, int total, int cshift) {
  int i = blockIdx.x * blockDim.x + threadIdx.x;
  if (i >= total) return;
  float dg = deg[i >> cshift];
  dg = dg < 1.0f ? 1.0f : dg;
  msg[i] = msg[i] / dg + xr[i];
}

// ------------------------------ BatchNorm ----------------------------------
__global__ void k_bn_stats(const float* __restrict__ X, int C,
                           float* __restrict__ mean, float* __restrict__ inv) {
  int c = blockIdx.x;
  float s = 0.f, s2 = 0.f;
  for (int rIdx = threadIdx.x; rIdx < NNODES; rIdx += 256) {
    float v = X[(size_t)rIdx * C + c];
    s += v; s2 += v * v;
  }
  __shared__ float sh_s[256], sh_q[256];
  sh_s[threadIdx.x] = s; sh_q[threadIdx.x] = s2;
  __syncthreads();
  for (int st = 128; st > 0; st >>= 1) {
    if (threadIdx.x < st) {
      sh_s[threadIdx.x] += sh_s[threadIdx.x + st];
      sh_q[threadIdx.x] += sh_q[threadIdx.x + st];
    }
    __syncthreads();
  }
  if (threadIdx.x == 0) {
    float m = sh_s[0] * (1.0f / NNODES);
    float var = sh_q[0] * (1.0f / NNODES) - m * m;
    if (var < 0.f) var = 0.f;
    mean[c] = m;
    inv[c]  = rsqrtf(var + 1e-5f);
  }
}

__global__ void k_bn_elu(const float* __restrict__ X, float* __restrict__ Y,
                         int total, int cmask,
                         const float* __restrict__ mean, const float* __restrict__ inv,
                         const float* __restrict__ g, const float* __restrict__ b) {
  int i = blockIdx.x * blockDim.x + threadIdx.x;
  if (i >= total) return;
  int c = i & cmask;
  float v = (X[i] - mean[c]) * inv[c] * g[c] + b[c];
  Y[i] = v > 0.f ? v : expm1f(v);
}

// ------------------------------ GAT attention ------------------------------
// per-node head logits: asrc/adst[n][h] = dot(xh[n,h,:], a_s/a_d[h,:]), d=128
__global__ void k_att_node(const float* __restrict__ xh,
                           const float* __restrict__ a_s, const float* __restrict__ a_d,
                           float* __restrict__ asrc, float* __restrict__ adst) {
  int n = blockIdx.x * 8 + (threadIdx.x >> 5);
  if (n >= NNODES) return;
  int lane = threadIdx.x & 31;
  const float* row = xh + (size_t)n * 512;
#pragma unroll
  for (int h = 0; h < 4; ++h) {
    float ps = 0.f, pd = 0.f;
#pragma unroll
    for (int i = 0; i < 4; ++i) {
      int d = lane + i * 32;
      float v = row[h * 128 + d];
      ps += v * a_s[h * 128 + d];
      pd += v * a_d[h * 128 + d];
    }
    for (int m = 16; m > 0; m >>= 1) {
      ps += __shfl_xor(ps, m, 32);
      pd += __shfl_xor(pd, m, 32);
    }
    if (lane == 0) { asrc[n * 4 + h] = ps; adst[n * 4 + h] = pd; }
  }
}

// leaky-relu logits + segment max over dst (self-loops appended: e >= E)
__global__ void k_att_logits(const int* __restrict__ src, const int* __restrict__ dst,
                             const float* __restrict__ asrc, const float* __restrict__ adst,
                             float* __restrict__ logit, float* __restrict__ mx) {
  int i = blockIdx.x * blockDim.x + threadIdx.x;
  if (i >= (NEDGES + NNODES) * 4) return;
  int e = i >> 2, h = i & 3;
  int s, d;
  if (e < NEDGES) { s = src[e]; d = dst[e]; } else { s = d = e - NEDGES; }
  float v = asrc[s * 4 + h] + adst[d * 4 + h];
  v = v > 0.f ? v : 0.2f * v;
  logit[i] = v;
  atomMaxF(&mx[d * 4 + h], v);
}

__global__ void k_att_exp(const int* __restrict__ dst, const float* __restrict__ logit,
                          const float* __restrict__ mx, float* __restrict__ exv,
                          float* __restrict__ den) {
  int i = blockIdx.x * blockDim.x + threadIdx.x;
  if (i >= (NEDGES + NNODES) * 4) return;
  int e = i >> 2, h = i & 3;
  int d = e < NEDGES ? dst[e] : e - NEDGES;
  float ex = __expf(logit[i] - mx[d * 4 + h]);
  exv[i] = ex;
  atomAddF(&den[d * 4 + h], ex);
}

// out[dst] += alpha * xh[src], 512 features, one wave per edge
__global__ void k_att_scatter(const int* __restrict__ src, const int* __restrict__ dst,
                              const float* __restrict__ exv, const float* __restrict__ den,
                              const float* __restrict__ xh, float* __restrict__ out) {
  int e = blockIdx.x * 8 + (threadIdx.x >> 5);
  if (e >= NEDGES + NNODES) return;
  int lane = threadIdx.x & 31;
  int s, d;
  if (e < NEDGES) { s = src[e]; d = dst[e]; } else { s = d = e - NEDGES; }
  float a0 = exv[e * 4 + 0] / den[d * 4 + 0];
  float a1 = exv[e * 4 + 1] / den[d * 4 + 1];
  float a2 = exv[e * 4 + 2] / den[d * 4 + 2];
  float a3 = exv[e * 4 + 3] / den[d * 4 + 3];
  const float* xr = xh + (size_t)s * 512;
  float* orow = out + (size_t)d * 512;
#pragma unroll
  for (int j = 0; j < 16; ++j) {
    int f = lane + j * 32;
    float al = (j < 4) ? a0 : (j < 8) ? a1 : (j < 12) ? a2 : a3;
    atomAddF(&orow[f], al * xr[f]);
  }
}

// ------------------------------ final FC2 ----------------------------------
__global__ void k_fc2(const float* __restrict__ h4, const float* __restrict__ W,
                      const float* __restrict__ b, float* __restrict__ out) {
  int n = blockIdx.x * blockDim.x + threadIdx.x;
  if (n >= NNODES) return;
  float a0 = b[0], a1 = b[1];
  const float* r = h4 + (size_t)n * 64;
#pragma unroll 8
  for (int k = 0; k < 64; ++k) {
    float v = r[k];
    a0 += v * W[k];
    a1 += v * W[64 + k];
  }
  out[n * 2 + 0] = a0;
  out[n * 2 + 1] = a1;
}

// ---------------------------------------------------------------------------
static inline unsigned cdiv(unsigned a, unsigned b) { return (a + b - 1) / b; }

extern "C" void kernel_launch(void* const* d_in, const int* in_sizes, int n_in,
                              void* d_out, int out_size, void* d_ws, size_t ws_size,
                              hipStream_t stream) {
  (void)in_sizes; (void)n_in; (void)out_size; (void)ws_size;
  const int N = NNODES, E = NEDGES;

  const float* x    = (const float*)d_in[0];
  const int*   ei   = (const int*)d_in[1];
  const int*   src  = ei;
  const int*   dst  = ei + E;
  const float* W1l  = (const float*)d_in[2];
  const float* W1r  = (const float*)d_in[4];
  const float* g1   = (const float*)d_in[5];
  const float* be1  = (const float*)d_in[6];
  const float* Wg   = (const float*)d_in[7];
  const float* a_s  = (const float*)d_in[8];
  const float* a_d  = (const float*)d_in[9];
  const float* g2   = (const float*)d_in[11];
  const float* be2  = (const float*)d_in[12];
  const float* W2l  = (const float*)d_in[13];
  const float* W2r  = (const float*)d_in[15];
  const float* g3   = (const float*)d_in[16];
  const float* be3  = (const float*)d_in[17];
  const float* Wf1  = (const float*)d_in[18];
  const float* g4   = (const float*)d_in[20];
  const float* be4  = (const float*)d_in[21];
  const float* Wf2  = (const float*)d_in[22];
  const float* bf2  = (const float*)d_in[23];
  float* out = (float*)d_out;

  // -------- workspace layout (all 256B aligned; ~430 MB total) --------
  char* ws = (char*)d_ws;
  size_t off = 0;
  auto alloc = [&](size_t bytes) -> char* {
    char* p = ws + off;
    off += (bytes + 255) & ~(size_t)255;
    return p;
  };
  unsigned short* xb   = (unsigned short*)alloc((size_t)N * 256 * 2);
  unsigned short* w1lb = (unsigned short*)alloc(128 * 256 * 2);
  unsigned short* w1rb = (unsigned short*)alloc(128 * 256 * 2);
  unsigned short* wgb  = (unsigned short*)alloc(512 * 128 * 2);
  unsigned short* w2lb = (unsigned short*)alloc(128 * 512 * 2);
  unsigned short* w2rb = (unsigned short*)alloc(128 * 512 * 2);
  unsigned short* wf1b = (unsigned short*)alloc(64 * 128 * 2);
  float* xl   = (float*)alloc((size_t)N * 128 * 4);   // also FC1 output (N x 64)
  float* xr   = (float*)alloc((size_t)N * 128 * 4);
  float* msg  = (float*)alloc((size_t)N * 128 * 4);   // sage sum -> combine in place
  float* deg  = (float*)alloc((size_t)N * 4);
  float* h1   = (float*)alloc((size_t)N * 128 * 4);   // layer1 / layer3 activations
  unsigned short* h1b = (unsigned short*)alloc((size_t)N * 128 * 2);
  float* xh   = (float*)alloc((size_t)N * 512 * 4);   // GAT per-head features
  float* gao  = (float*)alloc((size_t)N * 512 * 4);   // GAT out -> h2 in place
  unsigned short* h2b = (unsigned short*)alloc((size_t)N * 512 * 2);
  float* asrc = (float*)alloc((size_t)N * 4 * 4);
  float* adst = (float*)alloc((size_t)N * 4 * 4);
  float* mx   = (float*)alloc((size_t)N * 4 * 4);
  float* den  = (float*)alloc((size_t)N * 4 * 4);
  float* exv  = (float*)alloc((size_t)(E + N) * 4 * 4);
  float* h4   = (float*)alloc((size_t)N * 64 * 4);
  float* bmean = (float*)alloc(512 * 4);
  float* binv  = (float*)alloc(512 * 4);

  auto cvt = [&](const float* in, unsigned short* o, int n) {
    k_cvt_bf16<<<cdiv(n, 256), 256, 0, stream>>>(in, o, n);
  };
  auto gemm = [&](const unsigned short* A, const unsigned short* W, float* C,
                  int K, int Ncols) {
    int nt64 = Ncols / 64;
    int total = (N / 16) * nt64;   // N/16 == 3125 exactly
    k_gemm_bf16<<<cdiv(total, 8), 256, 0, stream>>>(A, W, C, K, Ncols, total, nt64);
  };

  // -------- weight + input conversions --------
  cvt(x,   xb,   N * 256);
  cvt(W1l, w1lb, 128 * 256);
  cvt(W1r, w1rb, 128 * 256);
  cvt(Wg,  wgb,  512 * 128);
  cvt(W2l, w2lb, 128 * 512);
  cvt(W2r, w2rb, 128 * 512);
  cvt(Wf1, wf1b, 64 * 128);

  // -------- in-degree --------
  hipMemsetAsync(deg, 0, (size_t)N * 4, stream);
  k_deg<<<cdiv(E, 256), 256, 0, stream>>>(dst, deg);

  // -------- layer 1: SAGE(256 -> 128), BN, ELU --------
  gemm(xb, w1lb, xl, 256, 128);
  gemm(xb, w1rb, xr, 256, 128);
  hipMemsetAsync(msg, 0, (size_t)N * 128 * 4, stream);
  k_sage_agg<<<cdiv(E, 8), 256, 0, stream>>>(src, dst, xl, msg);
  k_sage_combine<<<cdiv(N * 128, 256), 256, 0, stream>>>(msg, xr, deg, N * 128, 7);
  k_bn_stats<<<128, 256, 0, stream>>>(msg, 128, bmean, binv);
  k_bn_elu<<<cdiv(N * 128, 256), 256, 0, stream>>>(msg, h1, N * 128, 127, bmean, binv, g1, be1);

  // -------- layer 2: GAT(128 -> 4x128), BN, ELU --------
  cvt(h1, h1b, N * 128);
  gemm(h1b, wgb, xh, 128, 512);
  k_att_node<<<cdiv(N, 8), 256, 0, stream>>>(xh, a_s, a_d, asrc, adst);
  k_fill_f32<<<cdiv(N * 4, 256), 256, 0, stream>>>(mx, -INFINITY, N * 4);
  hipMemsetAsync(den, 0, (size_t)N * 4 * 4, stream);
  hipMemsetAsync(gao, 0, (size_t)N * 512 * 4, stream);
  k_att_logits<<<cdiv((E + N) * 4, 256), 256, 0, stream>>>(src, dst, asrc, adst, exv, mx);
  // exv currently holds raw logits; k_att_exp rewrites it with exp values.
  k_att_exp<<<cdiv((E + N) * 4, 256), 256, 0, stream>>>(dst, exv, mx, exv, den);
  k_att_scatter<<<cdiv(E + N, 8), 256, 0, stream>>>(src, dst, exv, den, xh, gao);
  k_bn_stats<<<512, 256, 0, stream>>>(gao, 512, bmean, binv);
  k_bn_elu<<<cdiv(N * 512, 256), 256, 0, stream>>>(gao, gao, N * 512, 511, bmean, binv, g2, be2);

  // -------- layer 3: SAGE(512 -> 128), BN, ELU --------
  cvt(gao, h2b, N * 512);
  gemm(h2b, w2lb, xl, 512, 128);
  gemm(h2b, w2rb, xr, 512, 128);
  hipMemsetAsync(msg, 0, (size_t)N * 128 * 4, stream);
  k_sage_agg<<<cdiv(E, 8), 256, 0, stream>>>(src, dst, xl, msg);
  k_sage_combine<<<cdiv(N * 128, 256), 256, 0, stream>>>(msg, xr, deg, N * 128, 7);
  k_bn_stats<<<128, 256, 0, stream>>>(msg, 128, bmean, binv);
  k_bn_elu<<<cdiv(N * 128, 256), 256, 0, stream>>>(msg, h1, N * 128, 127, bmean, binv, g3, be3);

  // -------- FC1(128 -> 64), BN, ELU --------
  cvt(h1, h1b, N * 128);
  gemm(h1b, wf1b, xl, 128, 64);   // xl reused as N x 64
  k_bn_stats<<<64, 256, 0, stream>>>(xl, 64, bmean, binv);
  k_bn_elu<<<cdiv(N * 64, 256), 256, 0, stream>>>(xl, h4, N * 64, 63, bmean, binv, g4, be4);

  // -------- FC2(64 -> 2) --------
  k_fc2<<<cdiv(N, 256), 256, 0, stream>>>(h4, Wf2, bf2, out);
}
